// Attention_Module_15152644620833
// MI455X (gfx1250) — matrix operand order repaired and checked by reading, not compiled, not run on hardware
//
#include <hip/hip_runtime.h>

typedef unsigned short u16;
typedef unsigned int   u32;

typedef __attribute__((ext_vector_type(16))) __bf16        bf16x16;
typedef __attribute__((ext_vector_type(8)))  float         f32x8;
typedef __attribute__((ext_vector_type(8)))  unsigned short u16x8;

union FragBF {
    bf16x16 bf;
    u16x8   h[2];
};

#define T_DIM 4096
#define B_DIM 8
#define D_DIM 1024
#define H_DIM 64
#define HALF  2048

__device__ __forceinline__ u16 f2bfu(float f) {
    u32 u = __float_as_uint(f);
    u32 r = u + 0x7fffu + ((u >> 16) & 1u);   // round-to-nearest-even
    return (u16)(r >> 16);
}

__device__ __forceinline__ f32x8 zero8() {
    f32x8 z = {0.f, 0.f, 0.f, 0.f, 0.f, 0.f, 0.f, 0.f};
    return z;
}

__device__ __forceinline__ f32x8 wmma_bf16(const FragBF& a, const FragBF& b, f32x8 c) {
    return __builtin_amdgcn_wmma_f32_16x16x32_bf16(false, a.bf, false, b.bf,
                                                   (short)0, c, false, false);
}

// ---------------------------------------------------------------------------
// Phase 0: conversions / precompute
// ---------------------------------------------------------------------------
__global__ void txl_convert_kernel(const float* __restrict__ x, u16* __restrict__ y, long n) {
    for (long i = (long)blockIdx.x * blockDim.x + threadIdx.x; i < n;
         i += (long)gridDim.x * blockDim.x)
        y[i] = f2bfu(x[i]);
}

__global__ void txl_pe_kernel(u16* __restrict__ pe) {
    int idx = blockIdx.x * blockDim.x + threadIdx.x;
    if (idx >= T_DIM * (D_DIM / 2)) return;
    int t = idx / (D_DIM / 2);
    int i = idx % (D_DIM / 2);
    float freq = __expf(-(float)(2 * i) * (9.2103403719761836f / (float)D_DIM));
    float ang  = (float)t * freq;
    pe[(size_t)t * D_DIM + 2 * i]     = f2bfu(__sinf(ang));
    pe[(size_t)t * D_DIM + 2 * i + 1] = f2bfu(__cosf(ang));
}

// W (D x H) f32 -> Wt (H x D) bf16
__global__ void txl_trw_kernel(const float* __restrict__ w, u16* __restrict__ wt) {
    int idx = blockIdx.x * blockDim.x + threadIdx.x;
    if (idx >= H_DIM * D_DIM) return;
    int h = idx >> 10;            // / D_DIM
    int k = idx & (D_DIM - 1);
    wt[(size_t)h * D_DIM + k] = f2bfu(w[(size_t)k * H_DIM + h]);
}

__global__ void txl_sums_kernel(const float* __restrict__ u, const float* __restrict__ v,
                                float* __restrict__ sums) {
    __shared__ float su[H_DIM], sv[H_DIM];
    int i = threadIdx.x;
    su[i] = u[i];
    sv[i] = v[i];
    __syncthreads();
    if (i == 0) {
        float a = 0.f, b = 0.f;
        for (int j = 0; j < H_DIM; ++j) { a += su[j]; b += sv[j]; }
        sums[0] = a;   // sum(u)
        sums[1] = b;   // sum(v)
    }
}

// ---------------------------------------------------------------------------
// Phase 1: R = PE @ Wr + br   (f32 out), vr[m] = sum(v) * sum_h R[m,h]
// grid = T/16 blocks, 128 threads (4 waves, one 16-col tile each)
// ---------------------------------------------------------------------------
__global__ void __launch_bounds__(128)
txl_rproj_kernel(const u16* __restrict__ PEbf, const u16* __restrict__ WrT,
                 const float* __restrict__ br, const float* __restrict__ sums,
                 float* __restrict__ Rf, float* __restrict__ vrW) {
    __shared__ float rsums[16];
    const int warp = threadIdx.x >> 5;
    const int lane = threadIdx.x & 31;
    const int col  = lane & 15;
    const int g    = lane >> 4;
    const int t0   = blockIdx.x * 16;
    const int n0   = warp * 16;

    if (threadIdx.x < 16) rsums[threadIdx.x] = 0.f;
    __syncthreads();

    f32x8 acc = zero8();
    const u16* arow = PEbf + (size_t)(t0 + col) * D_DIM;
    const u16* brow = WrT  + (size_t)(n0 + col) * D_DIM;
    for (int kk = 0; kk < D_DIM; kk += 32) {
        FragBF a, b;
        a.h[0] = *(const u16x8*)(arow + kk + 16 * g);
        a.h[1] = *(const u16x8*)(arow + kk + 16 * g + 8);
        b.h[0] = *(const u16x8*)(brow + kk + 16 * g);
        b.h[1] = *(const u16x8*)(brow + kk + 16 * g + 8);
        acc = wmma_bf16(a, b, acc);
    }

    const float brc = br[n0 + col];
    float rp[8];
#pragma unroll
    for (int r = 0; r < 8; ++r) {
        int   t  = t0 + r + 8 * g;
        float rv = acc[r] + brc;
        Rf[(size_t)t * H_DIM + n0 + col] = rv;
        rp[r] = rv;
    }
#pragma unroll
    for (int off = 1; off < 16; off <<= 1)
#pragma unroll
        for (int r = 0; r < 8; ++r) rp[r] += __shfl_xor(rp[r], off, 32);
    if (col == 0) {
#pragma unroll
        for (int r = 0; r < 8; ++r) atomicAdd(&rsums[r + 8 * g], rp[r]);
    }
    __syncthreads();
    if (threadIdx.x < 16) vrW[t0 + threadIdx.x] = sums[1] * rsums[threadIdx.x];
}

// ---------------------------------------------------------------------------
// Phase 2: projections q,k,v.  Rows of X are (t,b) pairs (row = t*B + b).
//   K2[b][t][h] = k + R[t]      (bf16)
//   Vt[b][h][t] = v             (bf16, transposed for P@V B-fragments)
//   Q [b][t][h] = q             (bf16, only t < HALF)
//   bias[b][t]  = sum(u)*sum_h k + vr[t]
// grid = T*B/16 blocks, 128 threads
// ---------------------------------------------------------------------------
__global__ void __launch_bounds__(128)
txl_proj_kernel(const u16* __restrict__ Xbf,
                const u16* __restrict__ WqT, const u16* __restrict__ WkT,
                const u16* __restrict__ WvT,
                const float* __restrict__ bq, const float* __restrict__ bk,
                const float* __restrict__ bv,
                const float* __restrict__ Rf, const float* __restrict__ vrW,
                const float* __restrict__ sums,
                u16* __restrict__ Qbf, u16* __restrict__ K2bf, u16* __restrict__ Vtbf,
                float* __restrict__ biasW) {
    __shared__ float ksums[16];
    const int  warp = threadIdx.x >> 5;
    const int  lane = threadIdx.x & 31;
    const int  col  = lane & 15;
    const int  g    = lane >> 4;
    const int  r0   = blockIdx.x * 16;
    const int  n0   = warp * 16;
    const bool doQ  = r0 < HALF * B_DIM;   // uniform over block: EXEC all-ones for WMMA

    if (threadIdx.x < 16) ksums[threadIdx.x] = 0.f;
    __syncthreads();

    f32x8 ak = zero8(), av = zero8(), aq = zero8();
    const u16* xrow = Xbf + (size_t)(r0 + col) * D_DIM;
    const u16* wkr  = WkT + (size_t)(n0 + col) * D_DIM;
    const u16* wvr  = WvT + (size_t)(n0 + col) * D_DIM;
    const u16* wqr  = WqT + (size_t)(n0 + col) * D_DIM;

    for (int kk = 0; kk < D_DIM; kk += 32) {
        FragBF a, b;
        a.h[0] = *(const u16x8*)(xrow + kk + 16 * g);
        a.h[1] = *(const u16x8*)(xrow + kk + 16 * g + 8);
        b.h[0] = *(const u16x8*)(wkr + kk + 16 * g);
        b.h[1] = *(const u16x8*)(wkr + kk + 16 * g + 8);
        ak = wmma_bf16(a, b, ak);
        b.h[0] = *(const u16x8*)(wvr + kk + 16 * g);
        b.h[1] = *(const u16x8*)(wvr + kk + 16 * g + 8);
        av = wmma_bf16(a, b, av);
        if (doQ) {
            b.h[0] = *(const u16x8*)(wqr + kk + 16 * g);
            b.h[1] = *(const u16x8*)(wqr + kk + 16 * g + 8);
            aq = wmma_bf16(a, b, aq);
        }
    }

    const float bkc = bk[n0 + col], bvc = bv[n0 + col], bqc = bq[n0 + col];
    float kp[8];
#pragma unroll
    for (int r = 0; r < 8; ++r) {
        int   row  = r0 + r + 8 * g;
        int   t    = row >> 3;          // / B_DIM
        int   b    = row & 7;
        float kval = ak[r] + bkc;
        kp[r]      = kval;
        K2bf[((size_t)b * T_DIM + t) * H_DIM + n0 + col] =
            f2bfu(kval + Rf[(size_t)t * H_DIM + n0 + col]);
        Vtbf[((size_t)b * H_DIM + n0 + col) * T_DIM + t] = f2bfu(av[r] + bvc);
        if (doQ)
            Qbf[((size_t)b * HALF + t) * H_DIM + n0 + col] = f2bfu(aq[r] + bqc);
    }
#pragma unroll
    for (int off = 1; off < 16; off <<= 1)
#pragma unroll
        for (int r = 0; r < 8; ++r) kp[r] += __shfl_xor(kp[r], off, 32);
    if (col == 0) {
#pragma unroll
        for (int r = 0; r < 8; ++r) atomicAdd(&ksums[r + 8 * g], kp[r]);
    }
    __syncthreads();
    if (threadIdx.x < 16) {
        int row = r0 + threadIdx.x;
        int t   = row >> 3;
        int b   = row & 7;
        biasW[(size_t)b * T_DIM + t] = sums[0] * ksums[threadIdx.x] + vrW[t];
    }
}

// ---------------------------------------------------------------------------
// Phase 3: flash attention.  One wave = one (b, 16-query tile).
// grid = B*HALF/16/4 blocks, 128 threads.
// ---------------------------------------------------------------------------
__global__ void __launch_bounds__(128)
txl_attn_kernel(const u16* __restrict__ Qbf, const u16* __restrict__ K2bf,
                const u16* __restrict__ Vtbf, const float* __restrict__ biasW,
                float* __restrict__ out) {
    __shared__ u16 pstage[4][16 * 32];
    const int warp = threadIdx.x >> 5;
    const int lane = threadIdx.x & 31;
    const int col  = lane & 15;
    const int g    = lane >> 4;
    const int wid  = blockIdx.x * 4 + warp;
    const int b    = wid >> 7;             // / (HALF/16)
    const int t0   = (wid & 127) * 16;

    // Q A-fragments: 16x64 = two K=32 slabs
    FragBF qa0, qa1;
    {
        const u16* qrow = Qbf + ((size_t)b * HALF + t0 + col) * H_DIM;
        qa0.h[0] = *(const u16x8*)(qrow + 16 * g);
        qa0.h[1] = *(const u16x8*)(qrow + 16 * g + 8);
        qa1.h[0] = *(const u16x8*)(qrow + 32 + 16 * g);
        qa1.h[1] = *(const u16x8*)(qrow + 32 + 16 * g + 8);
    }

    f32x8 acc0 = zero8(), acc1 = zero8(), acc2 = zero8(), acc3 = zero8();
    float mi[8], li[8];
#pragma unroll
    for (int r = 0; r < 8; ++r) { mi[r] = -1e30f; li[r] = 0.f; }

    u16* myp = pstage[warp];

    for (int m0 = 0; m0 < T_DIM; m0 += 32) {
        if (m0 > t0 + 15 && m0 + 31 < HALF) continue;   // fully masked tile

        // S = Q @ K2^T  (16 queries x 32 keys), split into two 16x16 C tiles
        f32x8 s0 = zero8(), s1 = zero8();
        {
            FragBF kb;
            const u16* kr0 = K2bf + ((size_t)b * T_DIM + m0 + col) * H_DIM;
            kb.h[0] = *(const u16x8*)(kr0 + 16 * g);
            kb.h[1] = *(const u16x8*)(kr0 + 16 * g + 8);
            s0 = wmma_bf16(qa0, kb, s0);
            kb.h[0] = *(const u16x8*)(kr0 + 32 + 16 * g);
            kb.h[1] = *(const u16x8*)(kr0 + 32 + 16 * g + 8);
            s0 = wmma_bf16(qa1, kb, s0);
            const u16* kr1 = K2bf + ((size_t)b * T_DIM + m0 + 16 + col) * H_DIM;
            kb.h[0] = *(const u16x8*)(kr1 + 16 * g);
            kb.h[1] = *(const u16x8*)(kr1 + 16 * g + 8);
            s1 = wmma_bf16(qa0, kb, s1);
            kb.h[0] = *(const u16x8*)(kr1 + 32 + 16 * g);
            kb.h[1] = *(const u16x8*)(kr1 + 32 + 16 * g + 8);
            s1 = wmma_bf16(qa1, kb, s1);
        }

        // bias (constant over query rows) + causal mask
        const float bias0 = biasW[(size_t)b * T_DIM + m0 + col];
        const float bias1 = biasW[(size_t)b * T_DIM + m0 + 16 + col];
        const int   key0  = m0 + col;
        const int   key1  = m0 + 16 + col;
        float rmax[8], rsum[8], p0[8], p1[8];
#pragma unroll
        for (int r = 0; r < 8; ++r) {
            int   qrow = t0 + r + 8 * g;
            float v0   = s0[r] + bias0;
            float v1   = s1[r] + bias1;
            if (key0 < HALF && key0 > qrow) v0 = -1e20f;
            if (key1 < HALF && key1 > qrow) v1 = -1e20f;
            s0[r]   = v0;
            s1[r]   = v1;
            rmax[r] = fmaxf(v0, v1);
        }
#pragma unroll
        for (int off = 1; off < 16; off <<= 1)
#pragma unroll
            for (int r = 0; r < 8; ++r) rmax[r] = fmaxf(rmax[r], __shfl_xor(rmax[r], off, 32));

#pragma unroll
        for (int r = 0; r < 8; ++r) {
            float mn = fmaxf(mi[r], rmax[r]);
            float sc = __expf(mi[r] - mn);
            p0[r]    = __expf(s0[r] - mn);
            p1[r]    = __expf(s1[r] - mn);
            rsum[r]  = p0[r] + p1[r];
            mi[r]    = mn;
            li[r] *= sc;
            acc0[r] *= sc; acc1[r] *= sc; acc2[r] *= sc; acc3[r] *= sc;
        }
#pragma unroll
        for (int off = 1; off < 16; off <<= 1)
#pragma unroll
            for (int r = 0; r < 8; ++r) rsum[r] += __shfl_xor(rsum[r], off, 32);
#pragma unroll
        for (int r = 0; r < 8; ++r) li[r] += rsum[r];

        // stage P through LDS: C layout -> A-fragment layout
#pragma unroll
        for (int r = 0; r < 8; ++r) {
            int row = r + 8 * g;
            myp[row * 32 + col]      = f2bfu(p0[r]);
            myp[row * 32 + 16 + col] = f2bfu(p1[r]);
        }
        asm volatile("s_wait_dscnt 0" ::: "memory");
        FragBF pa;
        pa.h[0] = *(const u16x8*)(&myp[col * 32 + 16 * g]);
        pa.h[1] = *(const u16x8*)(&myp[col * 32 + 16 * g + 8]);

        // O += P @ V  (B-fragments contiguous thanks to V^T layout)
        {
            FragBF vb;
            const u16* vc;
            vc = Vtbf + ((size_t)b * H_DIM + 0 + col) * T_DIM + m0 + 16 * g;
            vb.h[0] = *(const u16x8*)(vc);
            vb.h[1] = *(const u16x8*)(vc + 8);
            acc0 = wmma_bf16(pa, vb, acc0);
            vc = Vtbf + ((size_t)b * H_DIM + 16 + col) * T_DIM + m0 + 16 * g;
            vb.h[0] = *(const u16x8*)(vc);
            vb.h[1] = *(const u16x8*)(vc + 8);
            acc1 = wmma_bf16(pa, vb, acc1);
            vc = Vtbf + ((size_t)b * H_DIM + 32 + col) * T_DIM + m0 + 16 * g;
            vb.h[0] = *(const u16x8*)(vc);
            vb.h[1] = *(const u16x8*)(vc + 8);
            acc2 = wmma_bf16(pa, vb, acc2);
            vc = Vtbf + ((size_t)b * H_DIM + 48 + col) * T_DIM + m0 + 16 * g;
            vb.h[0] = *(const u16x8*)(vc);
            vb.h[1] = *(const u16x8*)(vc + 8);
            acc3 = wmma_bf16(pa, vb, acc3);
        }
    }

    // epilogue: normalize and store (half, B, H) f32
#pragma unroll
    for (int r = 0; r < 8; ++r) {
        int    t   = t0 + r + 8 * g;
        float  inv = 1.f / li[r];
        size_t base = ((size_t)t * B_DIM + b) * H_DIM + col;
        out[base + 0]  = acc0[r] * inv;
        out[base + 16] = acc1[r] * inv;
        out[base + 32] = acc2[r] * inv;
        out[base + 48] = acc3[r] * inv;
    }
}

// ---------------------------------------------------------------------------
extern "C" void kernel_launch(void* const* d_in, const int* in_sizes, int n_in,
                              void* d_out, int out_size, void* d_ws, size_t ws_size,
                              hipStream_t stream) {
    (void)in_sizes; (void)n_in; (void)out_size; (void)ws_size;

    const float* inp = (const float*)d_in[0];
    const float* Wq  = (const float*)d_in[1];
    const float* bq  = (const float*)d_in[2];
    const float* Wk  = (const float*)d_in[3];
    const float* bk  = (const float*)d_in[4];
    const float* Wv  = (const float*)d_in[5];
    const float* bv  = (const float*)d_in[6];
    const float* Wr  = (const float*)d_in[7];
    const float* br  = (const float*)d_in[8];
    const float* u   = (const float*)d_in[9];
    const float* v   = (const float*)d_in[10];
    float*       out = (float*)d_out;

    char*  ws  = (char*)d_ws;
    size_t off = 0;
    auto alloc = [&](size_t bytes) -> char* {
        char* p = ws + off;
        off = (off + bytes + 255) & ~(size_t)255;
        return p;
    };
    u16*   Xbf   = (u16*)alloc((size_t)T_DIM * B_DIM * D_DIM * 2);  // 67 MB
    u16*   PEbf  = (u16*)alloc((size_t)T_DIM * D_DIM * 2);          //  8 MB
    u16*   WqT   = (u16*)alloc((size_t)H_DIM * D_DIM * 2);
    u16*   WkT   = (u16*)alloc((size_t)H_DIM * D_DIM * 2);
    u16*   WvT   = (u16*)alloc((size_t)H_DIM * D_DIM * 2);
    u16*   WrT   = (u16*)alloc((size_t)H_DIM * D_DIM * 2);
    u16*   Qbf   = (u16*)alloc((size_t)B_DIM * HALF * H_DIM * 2);   //  2 MB
    u16*   K2bf  = (u16*)alloc((size_t)B_DIM * T_DIM * H_DIM * 2);  //  4 MB
    u16*   Vtbf  = (u16*)alloc((size_t)B_DIM * H_DIM * T_DIM * 2);  //  4 MB
    float* Rf    = (float*)alloc((size_t)T_DIM * H_DIM * 4);        //  1 MB
    float* vrW   = (float*)alloc((size_t)T_DIM * 4);
    float* biasW = (float*)alloc((size_t)B_DIM * T_DIM * 4);
    float* sumsW = (float*)alloc(64);

    const long nX = (long)T_DIM * B_DIM * D_DIM;
    txl_convert_kernel<<<4096, 256, 0, stream>>>(inp, Xbf, nX);
    txl_pe_kernel<<<(T_DIM * (D_DIM / 2) + 255) / 256, 256, 0, stream>>>(PEbf);
    txl_trw_kernel<<<(H_DIM * D_DIM + 255) / 256, 256, 0, stream>>>(Wq, WqT);
    txl_trw_kernel<<<(H_DIM * D_DIM + 255) / 256, 256, 0, stream>>>(Wk, WkT);
    txl_trw_kernel<<<(H_DIM * D_DIM + 255) / 256, 256, 0, stream>>>(Wv, WvT);
    txl_trw_kernel<<<(H_DIM * D_DIM + 255) / 256, 256, 0, stream>>>(Wr, WrT);
    txl_sums_kernel<<<1, 64, 0, stream>>>(u, v, sumsW);

    txl_rproj_kernel<<<T_DIM / 16, 128, 0, stream>>>(PEbf, WrT, br, sumsW, Rf, vrW);
    txl_proj_kernel<<<(T_DIM * B_DIM) / 16, 128, 0, stream>>>(
        Xbf, WqT, WkT, WvT, bq, bk, bv, Rf, vrW, sumsW, Qbf, K2bf, Vtbf, biasW);
    txl_attn_kernel<<<(B_DIM * HALF / 16) / 4, 128, 0, stream>>>(
        Qbf, K2bf, Vtbf, biasW, out);
}
